// GPT_68324339745059
// MI455X (gfx1250) — matrix operand order repaired and checked by reading, not compiled, not run on hardware
//
#include <hip/hip_runtime.h>

// ---------------- model dims ----------------
constexpr int kB    = 64;
constexpr int kT    = 256;
constexpr int kC    = 64;
constexpr int kNH   = 8;
constexpr int kHS   = 8;        // head size
constexpr int kL    = 10;
constexpr int kDFF  = 256;
constexpr int kV    = 8000;
constexpr int kM    = kB * kT;  // 16384 token rows
constexpr int kQKV  = 3 * kC;   // fused QKV width = 192

typedef __attribute__((ext_vector_type(16))) __bf16 v16bf;
typedef __attribute__((ext_vector_type(8)))  __bf16 v8bf;
typedef __attribute__((ext_vector_type(8)))  float  v8f;

// ---------------------------------------------------------------------------
// Tensor Data Mover staging (gfx1250 TDM), hedged per-toolchain.
// 2D tile: `rows` rows of `row_elems` bf16, global row stride
// `row_stride_elems`, LDS rows padded via TDM pad fields.
// ---------------------------------------------------------------------------
#if __has_builtin(__builtin_amdgcn_tensor_load_to_lds)
#define HAVE_TDM 1
typedef __attribute__((ext_vector_type(4))) unsigned int v4u;
typedef __attribute__((ext_vector_type(4))) int          v4i;
typedef __attribute__((ext_vector_type(8))) int          v8i;

__device__ __forceinline__ void tdm_load_2d(const void* gsrc, unsigned lds_off,
                                            unsigned rows, unsigned row_elems,
                                            unsigned row_stride_elems,
                                            unsigned pad_interval_code,
                                            unsigned pad_amount_code) {
  unsigned long long ga = (unsigned long long)gsrc;
  // group0: count=1 | lds_addr | global_addr[56:0] | type=2 ("image")
  v4u g0 = { 1u, lds_off, (unsigned)ga,
             (unsigned)((ga >> 32) & 0x1ffffffu) | (2u << 30) };
  // group1 word0: data_size=1 (2B), pad_enable, pad_interval, pad_amount
  unsigned w0 = (1u << 16) | (1u << 20) | (pad_interval_code << 22) |
                (pad_amount_code << 25);
  unsigned d1s = row_stride_elems * rows;       // tensor_dim1_stride
  v8i g1 = { (int)w0,
             (int)(row_elems << 16),            // tensor_dim0[15:0] @ bits63:48
             (int)(rows << 16),                 // tensor_dim1[15:0] @ bits95:80
             (int)(row_elems << 16),            // tile_dim0        @ bits127:112
             (int)rows,                         // tile_dim1        @ bits143:128
             (int)row_stride_elems,             // tensor_dim0_stride[31:0]
             (int)((d1s & 0xffffu) << 16),      // tensor_dim1_stride[15:0]
             (int)(d1s >> 16) };                // tensor_dim1_stride[47:16]
  v4i z4 = {0, 0, 0, 0};
#if __clang_major__ >= 23
  v8i z8 = {0, 0, 0, 0, 0, 0, 0, 0};
  __builtin_amdgcn_tensor_load_to_lds(g0, g1, z4, z4, z8, 0);
#else
  __builtin_amdgcn_tensor_load_to_lds(g0, g1, z4, z4, 0);
#endif
}
#else
#define HAVE_TDM 0
#endif

// ---------------------------------------------------------------------------
// x = tok_emb[idx] + pos_emb  (f32 residual stream)
// ---------------------------------------------------------------------------
__global__ void embed_kernel(const int* __restrict__ idx,
                             const float* __restrict__ tok,
                             const float* __restrict__ pos,
                             float* __restrict__ x) {
  int i = blockIdx.x * blockDim.x + threadIdx.x;   // over kM*kC
  int row = i >> 6;
  int c   = i & 63;
  int t   = row & (kT - 1);
  x[i] = tok[idx[row] * kC + c] + pos[t * kC + c];
}

// ---------------------------------------------------------------------------
// Pack a [L,K,Nloc] f32 weight into bf16 WMMA B-fragment order inside a
// packed matrix of total width 16*tilesNtot, at tile-column offset tnOff:
// out[l*K*Ntot + ((kb*tilesNtot + tnOff+tn)*32 + lane)*16 + j]
//   = W[l][kb*32 + (lane>>4)*8 + (j&7) + (j>>3)*16][tn*16 + (lane&15)]
// Each lane's 16 values for one (kblock, ntile) become one 32B load.
// ---------------------------------------------------------------------------
__global__ void pack_b_kernel(const float* __restrict__ W,
                              __bf16* __restrict__ out, int K, int Nloc,
                              int L, int tilesNtot, int tnOff) {
  int i = blockIdx.x * blockDim.x + threadIdx.x;
  if (i >= L * K * Nloc) return;
  int l = i / (K * Nloc);
  int p = i % (K * Nloc);
  int j    = p & 15;
  int lane = (p >> 4) & 31;
  int rest = p >> 9;
  int tilesNloc = Nloc >> 4;
  int tn = rest % tilesNloc;
  int kb = rest / tilesNloc;
  int kk  = kb * 32 + (lane >> 4) * 8 + (j & 7) + (j >> 3) * 16;
  int col = tn * 16 + (lane & 15);
  size_t oidx = (size_t)l * K * (16 * tilesNtot) +
                (((size_t)kb * tilesNtot + tnOff + tn) * 32 + lane) * 16 + j;
  out[oidx] = (__bf16)W[((size_t)l * K + kk) * Nloc + col];
}

// ---------------------------------------------------------------------------
// LayerNorm over C=64, one wave (32 lanes x 2 elems) per row, bf16 out
// ---------------------------------------------------------------------------
__global__ void ln_kernel(const float* __restrict__ x,
                          const float* __restrict__ g,
                          const float* __restrict__ b,
                          __bf16* __restrict__ out, int M) {
  int wave = (blockIdx.x * blockDim.x + threadIdx.x) >> 5;
  int lane = threadIdx.x & 31;
  if (wave >= M) return;
  const float* row = x + (size_t)wave * kC;
  float v0 = row[lane], v1 = row[lane + 32];
  float s = v0 + v1;
  #pragma unroll
  for (int off = 1; off < 32; off <<= 1) s += __shfl_xor(s, off, 32);
  float mu = s * (1.0f / kC);
  float d0 = v0 - mu, d1 = v1 - mu;
  float vs = d0 * d0 + d1 * d1;
  #pragma unroll
  for (int off = 1; off < 32; off <<= 1) vs += __shfl_xor(vs, off, 32);
  float rstd = rsqrtf(vs * (1.0f / kC) + 1e-5f);
  out[(size_t)wave * kC + lane]      = (__bf16)(d0 * rstd * g[lane] + b[lane]);
  out[(size_t)wave * kC + lane + 32] = (__bf16)(d1 * rstd * g[lane + 32] + b[lane + 32]);
}

// ---------------------------------------------------------------------------
// bf16 WMMA GEMM, wave computes a 16x64 strip (4 N-tiles, A-fragment reused
// across 4 back-to-back v_wmma). A row-major [M,K]; B pre-packed fragment
// order (pack_b_kernel). Fused bias / residual / relu / f32-or-bf16 output.
// Requires N % 64 == 0, K % 32 == 0.
// ---------------------------------------------------------------------------
__global__ void gemm_bf16_kernel(const __bf16* __restrict__ A,
                                 const __bf16* __restrict__ Bp,
                                 const float* __restrict__ bias,
                                 const float* __restrict__ residual,
                                 float* __restrict__ outF,
                                 __bf16* __restrict__ outB,
                                 int M, int N, int K, int relu) {
  int wave   = (blockIdx.x * blockDim.x + threadIdx.x) >> 5;
  int nStrip = N >> 6;
  int total  = (M >> 4) * nStrip;
  if (wave >= total) return;
  int tm = wave / nStrip;
  int ts = wave % nStrip;
  int lane = threadIdx.x & 31;
  int colg = lane & 15;
  int grp  = lane >> 4;
  int tilesN = N >> 4;

  const __bf16* Arow = A + ((size_t)(tm * 16 + colg)) * K + grp * 8;
  v8f acc[4] = {{}, {}, {}, {}};

  for (int kb = 0; kb < (K >> 5); ++kb) {
    // A fragment 16x32: two contiguous 16B runs per lane -> b128 loads
    v8bf alo = *(const v8bf*)(Arow + kb * 32);
    v8bf ahi = *(const v8bf*)(Arow + kb * 32 + 16);
    v16bf a = __builtin_shufflevector(alo, ahi, 0, 1, 2, 3, 4, 5, 6, 7,
                                      8, 9, 10, 11, 12, 13, 14, 15);
    __builtin_prefetch(Arow + kb * 32 + 32, 0, 1);
    #pragma unroll
    for (int t = 0; t < 4; ++t) {
      int tn = ts * 4 + t;
      v16bf bfrag = *(const v16bf*)(Bp +
          (((size_t)kb * tilesN + tn) * 32 + lane) * 16);   // 32B contiguous
      acc[t] = __builtin_amdgcn_wmma_f32_16x16x32_bf16(false, a, false, bfrag,
                                                       (short)0, acc[t],
                                                       false, false);
    }
  }

  #pragma unroll
  for (int t = 0; t < 4; ++t) {
    int gcol = (ts * 4 + t) * 16 + colg;
    float bi = bias ? bias[gcol] : 0.0f;
    #pragma unroll
    for (int j = 0; j < 8; ++j) {
      int grow = tm * 16 + grp * 8 + j;
      float val = acc[t][j] + bi;
      if (residual) val += residual[(size_t)grow * N + gcol];
      if (relu) val = fmaxf(val, 0.0f);
      if (outF) outF[(size_t)grow * N + gcol] = val;
      if (outB) outB[(size_t)grow * N + gcol] = (__bf16)val;
    }
  }
}

// ---------------------------------------------------------------------------
// Attention for one (batch, head) from the fused QKV activation [M, 192]:
// flash-style online softmax via WMMA. Q/K/V head slices staged in LDS
// (TDM async-tensor load where available; HS padded 8->32 so a score tile
// is one 16x16x32 WMMA). V kept column-major in LDS so the P@V B-fragment
// is one contiguous 32B ds load per lane.
// ---------------------------------------------------------------------------
__global__ void attn_kernel(const __bf16* __restrict__ qkv,
                            __bf16* __restrict__ o) {
  int b = blockIdx.x / kNH;
  int h = blockIdx.x % kNH;

  __shared__ __bf16 qs[kT][32];       // 16 KB, cols 8..31 zero
  __shared__ __bf16 ks[kT][32];       // 16 KB
  __shared__ __bf16 vsT[16][kT];      // 8 KB, V column-major, rows 8..15 zero
  __shared__ __bf16 ps[8][16][32];    // 8 KB, per-wave P scratch

  int tid = threadIdx.x;              // 256 threads = 8 waves
  const __bf16* qbase = qkv + ((size_t)b * kT) * kQKV + h * kHS;
  const __bf16* kbase = qbase + kC;
  const __bf16* vbase = qbase + 2 * kC;

#if HAVE_TDM
  // Async DMA the strided head slices into LDS (16B rows, 384B global
  // stride; TDM pad fields expand each LDS row of qs/ks to 64B).
  if (tid < 32) {
    tdm_load_2d(qbase, (unsigned)(size_t)&qs[0][0], kT, kHS, kQKV, /*4dw*/1, /*+12dw*/11);
    tdm_load_2d(kbase, (unsigned)(size_t)&ks[0][0], kT, kHS, kQKV, 1, 11);
  }
#if __has_builtin(__builtin_amdgcn_s_wait_tensorcnt)
  __builtin_amdgcn_s_wait_tensorcnt(0);
#endif
#endif

  for (int i = tid; i < kT * 32; i += 256) {
    int r = i >> 5, c = i & 31;
    __bf16 zv = (__bf16)0.0f;
    __bf16 qv = zv, kv = zv;
    if (c < kHS) {
      size_t off = (size_t)r * kQKV + c;
      qv = qbase[off];
      kv = kbase[off];
    }
    qs[r][c] = qv;
    ks[r][c] = kv;
  }
  for (int i = tid; i < kT * 16; i += 256) {
    int r = i >> 4, c = i & 15;
    vsT[c][r] = (c < kHS) ? vbase[(size_t)r * kQKV + c] : (__bf16)0.0f;
  }
  __syncthreads();

  int wave = tid >> 5;
  int lane = tid & 31;
  int colg = lane & 15;
  int grp  = lane >> 4;
  const float scale = 0.125f;         // 64^-0.5 (reference scales by C)

  for (int strip = wave; strip < kT / 16; strip += 8) {
    // Q strip A-fragment (16 rows x 32 padded K): two 16B LDS runs per lane
    v16bf aq;
    {
      int r = strip * 16 + colg;
      v8bf lo = *(const v8bf*)&qs[r][grp * 16];
      v8bf hi = *(const v8bf*)&qs[r][grp * 16 + 8];
      aq = __builtin_shufflevector(lo, hi, 0, 1, 2, 3, 4, 5, 6, 7,
                                   8, 9, 10, 11, 12, 13, 14, 15);
    }

    v8f acc = {};
    float m_i[8], l_i[8];
    #pragma unroll
    for (int j = 0; j < 8; ++j) { m_i[j] = -1e30f; l_i[j] = 0.0f; }

    int kpmax = (strip * 16 + 15) >> 5;        // causal: 32-key blocks needed
    for (int kp = 0; kp <= kpmax; ++kp) {
      v8f s[2];
      #pragma unroll
      for (int t2 = 0; t2 < 2; ++t2) {
        int kt = kp * 2 + t2;
        // B[kk][n] = K[kt*16+n][kk]: one contiguous 32B LDS run per lane
        v16bf bk = *(const v16bf*)&ks[kt * 16 + colg][grp * 16];
        v8f z = {};
        s[t2] = __builtin_amdgcn_wmma_f32_16x16x32_bf16(false, aq, false, bk,
                                                        (short)0, z, false, false);
      }
      float p0[8], p1[8];
      #pragma unroll
      for (int j = 0; j < 8; ++j) {
        int qrow = strip * 16 + grp * 8 + j;
        float v0 = s[0][j] * scale;
        float v1 = s[1][j] * scale;
        if (kp * 32 + colg      > qrow) v0 = -1e30f;
        if (kp * 32 + 16 + colg > qrow) v1 = -1e30f;
        float rmax = fmaxf(v0, v1);
        #pragma unroll
        for (int off = 1; off < 16; off <<= 1)
          rmax = fmaxf(rmax, __shfl_xor(rmax, off, 32));
        float mnew = fmaxf(m_i[j], rmax);
        float corr = __expf(m_i[j] - mnew);
        m_i[j] = mnew;
        p0[j] = __expf(v0 - mnew);
        p1[j] = __expf(v1 - mnew);
        float rsum = p0[j] + p1[j];
        #pragma unroll
        for (int off = 1; off < 16; off <<= 1)
          rsum += __shfl_xor(rsum, off, 32);
        l_i[j] = l_i[j] * corr + rsum;
        acc[j] *= corr;
      }
      // stage P (16x32) via wave-private LDS to re-swizzle C->A layout
      #pragma unroll
      for (int j = 0; j < 8; ++j) {
        int r = grp * 8 + j;
        ps[wave][r][colg]      = (__bf16)p0[j];
        ps[wave][r][16 + colg] = (__bf16)p1[j];
      }
      __builtin_amdgcn_wave_barrier();          // DS ops are wave-in-order
      v16bf ap;
      {
        v8bf lo = *(const v8bf*)&ps[wave][colg][grp * 16];
        v8bf hi = *(const v8bf*)&ps[wave][colg][grp * 16 + 8];
        ap = __builtin_shufflevector(lo, hi, 0, 1, 2, 3, 4, 5, 6, 7,
                                     8, 9, 10, 11, 12, 13, 14, 15);
      }
      // B[kk][n] = V[kp*32+kk][n]: vsT row -> one contiguous 32B LDS run
      v16bf bv = *(const v16bf*)&vsT[colg][kp * 32 + grp * 16];
      acc = __builtin_amdgcn_wmma_f32_16x16x32_bf16(false, ap, false, bv,
                                                    (short)0, acc, false, false);
    }

    if (colg < kHS) {
      #pragma unroll
      for (int j = 0; j < 8; ++j) {
        int trow = strip * 16 + grp * 8 + j;
        o[((size_t)b * kT + trow) * kC + h * kHS + colg] =
            (__bf16)(acc[j] / l_i[j]);
      }
    }
  }
}

// ---------------------------------------------------------------------------
// host launch
// ---------------------------------------------------------------------------
static inline int gemm_grid(int M, int N) {
  int strips = (M / 16) * (N / 64);
  return (strips + 7) / 8;            // 8 waves (256 threads) per block
}

extern "C" void kernel_launch(void* const* d_in, const int* in_sizes, int n_in,
                              void* d_out, int out_size, void* d_ws, size_t ws_size,
                              hipStream_t stream) {
  const int*   idx    = (const int*)  d_in[0];
  const float* tok    = (const float*)d_in[1];
  const float* pos    = (const float*)d_in[2];
  const float* wq     = (const float*)d_in[3];
  const float* wk     = (const float*)d_in[4];
  const float* wv     = (const float*)d_in[5];
  const float* wproj  = (const float*)d_in[6];
  const float* bproj  = (const float*)d_in[7];
  const float* ln1g   = (const float*)d_in[8];
  const float* ln1b   = (const float*)d_in[9];
  const float* ln2g   = (const float*)d_in[10];
  const float* ln2b   = (const float*)d_in[11];
  const float* w1     = (const float*)d_in[12];
  const float* b1     = (const float*)d_in[13];
  const float* w2     = (const float*)d_in[14];
  const float* b2     = (const float*)d_in[15];
  const float* lnfg   = (const float*)d_in[16];
  const float* lnfb   = (const float*)d_in[17];
  const float* whead  = (const float*)d_in[18];
  const float* bhead  = (const float*)d_in[19];

  char* cur = (char*)d_ws;
  auto alloc = [&](size_t bytes) {
    void* p = cur;
    cur += (bytes + 255) & ~(size_t)255;
    return p;
  };
  float*  X    = (float*) alloc((size_t)kM * kC * 4);
  __bf16* H    = (__bf16*)alloc((size_t)kM * kC * 2);
  __bf16* QKV  = (__bf16*)alloc((size_t)kM * kQKV * 2);
  __bf16* O    = (__bf16*)alloc((size_t)kM * kC * 2);
  __bf16* FF   = (__bf16*)alloc((size_t)kM * kDFF * 2);
  __bf16* Wqkv = (__bf16*)alloc((size_t)kL * kC * kQKV * 2);
  __bf16* Wpb  = (__bf16*)alloc((size_t)kL * kC * kC * 2);
  __bf16* W1b  = (__bf16*)alloc((size_t)kL * kC * kDFF * 2);
  __bf16* W2b  = (__bf16*)alloc((size_t)kL * kDFF * kC * 2);
  __bf16* Whb  = (__bf16*)alloc((size_t)kC * kV * 2);

  // pack all weights to bf16 fragment order (deterministic each call).
  // wq|wk|wv interleave into one packed matrix of width 192.
  int nqkv = kL * kC * kC;
  int nff  = kL * kC * kDFF;
  int nhd  = kC * kV;
  pack_b_kernel<<<(nqkv + 255) / 256, 256, 0, stream>>>(wq,    Wqkv, kC,   kC,   kL, kQKV / 16, 0);
  pack_b_kernel<<<(nqkv + 255) / 256, 256, 0, stream>>>(wk,    Wqkv, kC,   kC,   kL, kQKV / 16, 4);
  pack_b_kernel<<<(nqkv + 255) / 256, 256, 0, stream>>>(wv,    Wqkv, kC,   kC,   kL, kQKV / 16, 8);
  pack_b_kernel<<<(nqkv + 255) / 256, 256, 0, stream>>>(wproj, Wpb,  kC,   kC,   kL, kC / 16,   0);
  pack_b_kernel<<<(nff  + 255) / 256, 256, 0, stream>>>(w1,    W1b,  kC,   kDFF, kL, kDFF / 16, 0);
  pack_b_kernel<<<(nff  + 255) / 256, 256, 0, stream>>>(w2,    W2b,  kDFF, kC,   kL, kC / 16,   0);
  pack_b_kernel<<<(nhd  + 255) / 256, 256, 0, stream>>>(whead, Whb,  kC,   kV,   1,  kV / 16,   0);

  embed_kernel<<<(kM * kC) / 256, 256, 0, stream>>>(idx, tok, pos, X);

  for (int l = 0; l < kL; ++l) {
    ln_kernel<<<kM / 8, 256, 0, stream>>>(X, ln1g + l * kC, ln1b + l * kC, H, kM);

    // fused QKV: [16384,64] @ [64,192]
    gemm_bf16_kernel<<<gemm_grid(kM, kQKV), 256, 0, stream>>>(
        H, Wqkv + (size_t)l * kC * kQKV, nullptr, nullptr, nullptr, QKV,
        kM, kQKV, kC, 0);

    attn_kernel<<<kB * kNH, 256, 0, stream>>>(QKV, O);

    // x = x + O @ w_proj + b_proj  (f32 residual stream, in-place)
    gemm_bf16_kernel<<<gemm_grid(kM, kC), 256, 0, stream>>>(
        O, Wpb + (size_t)l * kC * kC, bproj + l * kC, X, X, nullptr, kM, kC, kC, 0);

    ln_kernel<<<kM / 8, 256, 0, stream>>>(X, ln2g + l * kC, ln2b + l * kC, H, kM);

    gemm_bf16_kernel<<<gemm_grid(kM, kDFF), 256, 0, stream>>>(
        H, W1b + (size_t)l * kC * kDFF, b1 + l * kDFF, nullptr, nullptr, FF,
        kM, kDFF, kC, 1);

    gemm_bf16_kernel<<<gemm_grid(kM, kC), 256, 0, stream>>>(
        FF, W2b + (size_t)l * kDFF * kC, b2 + l * kC, X, X, nullptr,
        kM, kC, kDFF, 0);
  }

  ln_kernel<<<kM / 8, 256, 0, stream>>>(X, lnfg, lnfb, H, kM);

  gemm_bf16_kernel<<<gemm_grid(kM, kV), 256, 0, stream>>>(
      H, Whb, bhead, nullptr, (float*)d_out, nullptr, kM, kV, kC, 0);
}
